// MultiheadAttention_54185307406968
// MI455X (gfx1250) — compile-verified, matrix operand order checked by reading, not run on hardware
//
#include <hip/hip_runtime.h>
#include <stdint.h>

// ---------------- problem constants ----------------
#define T_LEN 1024
#define BSZ   2
#define EMB   1024
#define NH    16
#define HD    64
#define MROWS (T_LEN * BSZ)   // 2048 rows for the E-GEMMs
#define SEQ   1024
#define QSCALE 0.125f         // 64^-0.5

// LDS-tiled GEMM geometry (proj/out kernels)
#define BM 128
#define BN 128
#define BK 32
#define LDSTRIDE 40           // padded row stride (elements) -> bank-conflict free

typedef __bf16 bf16_t;
typedef __attribute__((ext_vector_type(16))) __bf16 v16bf;
typedef __attribute__((ext_vector_type(8)))  __bf16 v8bf;
typedef __attribute__((ext_vector_type(8)))  float  v8f;
typedef __attribute__((ext_vector_type(4)))  float  v4f;

// ---------------- WMMA helper ----------------
__device__ __forceinline__ v8f wmma_bf16(v16bf a, v16bf b, v8f c) {
  return __builtin_amdgcn_wmma_f32_16x16x32_bf16(false, a, false, b, (short)0, c,
                                                 false, false);
}

// A/B fragment for 16x16x32 bf16 WMMA from a row-major, K-contiguous bf16
// matrix (global or LDS). p points at (row, koff), koff = (lane&16) ? 8 : 0.
// lane<16 holds K {0..7, 16..23}; lane>=16 holds K {8..15, 24..31}.
__device__ __forceinline__ v16bf frag_bf16(const bf16_t* __restrict__ p) {
  v8bf lo = *(const v8bf*)(p);       // 16B load (global_load_b128 / ds_load_b128)
  v8bf hi = *(const v8bf*)(p + 16);
  v16bf f;
#pragma unroll
  for (int i = 0; i < 8; ++i) { f[i] = lo[i]; f[i + 8] = hi[i]; }
  return f;
}

// Async-DMA one BMxBK (=128x32) bf16 tile, global -> LDS (padded rows).
// 512 x 16B transfers, 256 threads x 2 instructions; tracked by ASYNCcnt.
__device__ __forceinline__ void stage_tile_async(
    const bf16_t* __restrict__ gsrc, size_t ld, int kb,
    bf16_t* lbuf, int tid) {
#pragma unroll
  for (int j = 0; j < 2; ++j) {
    const int id  = tid + 256 * j;
    const int row = id >> 2;
    const int c8  = (id & 3) * 8;
    const uint64_t ga =
        (uint64_t)(uintptr_t)(gsrc + (size_t)row * ld + kb + c8);
    const uint32_t la = (uint32_t)(uintptr_t)(lbuf + row * LDSTRIDE + c8);
    asm volatile("global_load_async_to_lds_b128 %0, %1, off"
                 :: "v"(la), "v"(ga) : "memory");
  }
}

// ======================================================================
// Kernel 0: one-shot f32 -> bf16 conversion (query + the 4 weights).
// ======================================================================
__global__ __launch_bounds__(256) void conv_kernel(
    const float* __restrict__ src, bf16_t* __restrict__ dst) {
  const size_t i = ((size_t)blockIdx.x * 256 + threadIdx.x) * 8;
  v4f x0 = *(const v4f*)(src + i);
  v4f x1 = *(const v4f*)(src + i + 4);
  v8bf o;
#pragma unroll
  for (int j = 0; j < 4; ++j) {
    o[j]     = (__bf16)x0[j];
    o[j + 4] = (__bf16)x1[j];
  }
  *(v8bf*)(dst + i) = o;
}

// ======================================================================
// Kernel 1: projection  out = X @ W^T + b  (NT GEMM, bf16, LDS-tiled)
// Block tile 128x128, async-DMA double-buffered k-chunks of 32.
// 8 waves as 4(M) x 2(N); each wave: 32x64 register tile = 8 WMMAs/k-step
// from 6 LDS fragments.
//   mode 0: dst[b][h][t][d]   (Q with scale, K)
//   mode 1: dst[b][h][d][t]   (V transposed for the NT ctx GEMM)
// ======================================================================
__global__ __launch_bounds__(256) void proj_kernel(
    const bf16_t* __restrict__ X, const bf16_t* __restrict__ W,
    const float* __restrict__ bias, bf16_t* __restrict__ dst,
    float scale, int mode) {
  __shared__ __align__(16) bf16_t Asm[2][BM * LDSTRIDE];
  __shared__ __align__(16) bf16_t Bsm[2][BN * LDSTRIDE];

  const int tid  = threadIdx.x;
  const int lane = tid & 31;
  const int wave = tid >> 5;
  const int wm = wave >> 1;            // 0..3  (M sub-block of 32)
  const int wn = wave & 1;             // 0..1  (N sub-block of 64)
  const int bm = blockIdx.x >> 3;      // 16 M-blocks
  const int bn = blockIdx.x & 7;       // 8  N-blocks
  const int r16  = lane & 15;
  const int koff = (lane & 16) ? 8 : 0;

  const bf16_t* __restrict__ gA = X + (size_t)(bm * BM) * EMB;
  const bf16_t* __restrict__ gB = W + (size_t)(bn * BN) * EMB;

  v8f acc[2][4] = {};

  // prologue: stage chunk 0
  stage_tile_async(gA, EMB, 0, Asm[0], tid);
  stage_tile_async(gB, EMB, 0, Bsm[0], tid);

  const int NK = EMB / BK;             // 32 k-chunks
  for (int it = 0; it < NK; ++it) {
    const int cur = it & 1;
    if (it + 1 < NK) {
      stage_tile_async(gA, EMB, (it + 1) * BK, Asm[cur ^ 1], tid);
      stage_tile_async(gB, EMB, (it + 1) * BK, Bsm[cur ^ 1], tid);
      // allow the 4 just-issued DMAs to stay outstanding; drain older ones
      asm volatile("s_wait_asynccnt 0x4" ::: "memory");
    } else {
      asm volatile("s_wait_asynccnt 0x0" ::: "memory");
    }
    __syncthreads();

    const bf16_t* Ab = Asm[cur] + (wm * 32) * LDSTRIDE;
    const bf16_t* Bb = Bsm[cur] + (wn * 64) * LDSTRIDE;
    v16bf a0 = frag_bf16(Ab + (size_t)r16 * LDSTRIDE + koff);
    v16bf a1 = frag_bf16(Ab + (size_t)(16 + r16) * LDSTRIDE + koff);
    v16bf b0 = frag_bf16(Bb + (size_t)r16 * LDSTRIDE + koff);
    v16bf b1 = frag_bf16(Bb + (size_t)(16 + r16) * LDSTRIDE + koff);
    v16bf b2 = frag_bf16(Bb + (size_t)(32 + r16) * LDSTRIDE + koff);
    v16bf b3 = frag_bf16(Bb + (size_t)(48 + r16) * LDSTRIDE + koff);
    acc[0][0] = wmma_bf16(a0, b0, acc[0][0]);
    acc[0][1] = wmma_bf16(a0, b1, acc[0][1]);
    acc[0][2] = wmma_bf16(a0, b2, acc[0][2]);
    acc[0][3] = wmma_bf16(a0, b3, acc[0][3]);
    acc[1][0] = wmma_bf16(a1, b0, acc[1][0]);
    acc[1][1] = wmma_bf16(a1, b1, acc[1][1]);
    acc[1][2] = wmma_bf16(a1, b2, acc[1][2]);
    acc[1][3] = wmma_bf16(a1, b3, acc[1][3]);
    __syncthreads();  // all waves done reading before this buffer is re-staged
  }

#pragma unroll
  for (int ti = 0; ti < 2; ++ti) {
#pragma unroll
    for (int tj = 0; tj < 4; ++tj) {
      const int n  = bn * BN + wn * 64 + tj * 16 + r16;
      const int h  = n >> 6;
      const int d  = n & 63;
      const float bv = bias[n];
      const int mbase = bm * BM + wm * 32 + ti * 16 + ((lane >> 4) << 3);
#pragma unroll
      for (int i = 0; i < 8; ++i) {
        const int m  = mbase + i;
        const int t  = m >> 1;      // BSZ == 2
        const int bb = m & 1;
        const float v = (acc[ti][tj][i] + bv) * scale;
        size_t off;
        if (mode == 0)
          off = (((size_t)(bb * NH + h)) * T_LEN + t) * HD + d;   // [B,H,T,D]
        else
          off = (((size_t)(bb * NH + h)) * HD + d) * T_LEN + t;   // [B,H,D,T]
        dst[off] = (bf16_t)v;
      }
    }
  }
}

// ======================================================================
// Kernel 2: scores[b,h,t,s] = sum_d q[b,h,t,d] * k[b,h,s,d]  (NT, bf16)
// K=64 and operands are WGP$-resident: register-tiled 32x32 per wave.
// ======================================================================
__global__ __launch_bounds__(256) void scores_kernel(
    const bf16_t* __restrict__ qh, const bf16_t* __restrict__ kh,
    float* __restrict__ scores) {
  const int lane = threadIdx.x & 31;
  const int wave = threadIdx.x >> 5;
  const int gtile = blockIdx.x * 8 + wave;         // 32*1024 = 32768 tiles
  const int bh   = gtile >> 10;
  const int tile = gtile & 1023;
  const int tm = tile >> 5;
  const int tn = tile & 31;
  const int r16  = lane & 15;
  const int koff = (lane & 16) ? 8 : 0;

  const bf16_t* __restrict__ A = qh + (size_t)bh * T_LEN * HD;
  const bf16_t* __restrict__ B = kh + (size_t)bh * T_LEN * HD;
  const bf16_t* __restrict__ pa0 = A + (size_t)(tm * 32 + r16) * HD + koff;
  const bf16_t* __restrict__ pa1 = pa0 + (size_t)16 * HD;
  const bf16_t* __restrict__ pb0 = B + (size_t)(tn * 32 + r16) * HD + koff;
  const bf16_t* __restrict__ pb1 = pb0 + (size_t)16 * HD;

  v8f acc00 = {}, acc01 = {}, acc10 = {}, acc11 = {};
#pragma unroll
  for (int kb = 0; kb < HD; kb += 32) {
    v16bf a0 = frag_bf16(pa0 + kb);
    v16bf a1 = frag_bf16(pa1 + kb);
    v16bf b0 = frag_bf16(pb0 + kb);
    v16bf b1 = frag_bf16(pb1 + kb);
    acc00 = wmma_bf16(a0, b0, acc00);
    acc01 = wmma_bf16(a0, b1, acc01);
    acc10 = wmma_bf16(a1, b0, acc10);
    acc11 = wmma_bf16(a1, b1, acc11);
  }

  float* __restrict__ out = scores + (size_t)bh * T_LEN * SEQ;
  v8f accs[2][2] = {{acc00, acc01}, {acc10, acc11}};
#pragma unroll
  for (int ti = 0; ti < 2; ++ti) {
#pragma unroll
    for (int tj = 0; tj < 2; ++tj) {
      const int n = tn * 32 + tj * 16 + r16;
      const int mbase = tm * 32 + ti * 16 + ((lane >> 4) << 3);
#pragma unroll
      for (int i = 0; i < 8; ++i)
        out[(size_t)(mbase + i) * SEQ + n] = accs[ti][tj][i];
    }
  }
}

// ======================================================================
// Kernel 3: per-(b,t) softmax over S for all 16 heads + cross-head mix.
// Score rows staged into LDS with GLOBAL_LOAD_ASYNC_TO_LDS_B128 (ASYNCcnt).
// ======================================================================
__global__ __launch_bounds__(256) void softmax_mix_kernel(
    const float* __restrict__ scores, const float* __restrict__ mix_w,
    bf16_t* __restrict__ pmix) {
  extern __shared__ float smem[];
  float* sc = smem;                 // [16][1024]
  float* Wm = smem + NH * SEQ;      // [16][16]
  float* rl = Wm + NH * NH;         // [16] 1/sum_h

  const int bt = blockIdx.x;
  const int b  = bt >> 10;
  const int t  = bt & 1023;
  const float* __restrict__ src =
      scores + ((size_t)b * NH) * (size_t)T_LEN * SEQ + (size_t)t * SEQ;

  // stage all 16 head rows (64KB) into LDS via async DMA
  const uint32_t lds_base = (uint32_t)(uintptr_t)sc;
  for (int vi = threadIdx.x; vi < (NH * SEQ) / 4; vi += 256) {
    const int h  = vi >> 8;               // 256 float4 per row
    const int s4 = vi & 255;
    const uint64_t gaddr =
        (uint64_t)(uintptr_t)(src + (size_t)h * T_LEN * SEQ + (size_t)s4 * 4);
    const uint32_t laddr = lds_base + (uint32_t)vi * 16u;
    asm volatile("global_load_async_to_lds_b128 %0, %1, off"
                 :: "v"(laddr), "v"(gaddr) : "memory");
  }
  // column-stochastic softmax of mix_w (threads 0..15, one column each)
  if (threadIdx.x < NH) {
    const int g = threadIdx.x;
    float mx = -1e30f;
#pragma unroll
    for (int h = 0; h < NH; ++h) mx = fmaxf(mx, mix_w[h * NH + g]);
    float sum = 0.f;
#pragma unroll
    for (int h = 0; h < NH; ++h) sum += __expf(mix_w[h * NH + g] - mx);
    const float inv = 1.f / sum;
#pragma unroll
    for (int h = 0; h < NH; ++h)
      Wm[h * NH + g] = __expf(mix_w[h * NH + g] - mx) * inv;
  }
  asm volatile("s_wait_asynccnt 0x0" ::: "memory");
  __syncthreads();

  // per-head softmax stats; wave w owns heads w and w+8 (wave32 reductions)
  const int wave = threadIdx.x >> 5;
  const int lane = threadIdx.x & 31;
  for (int h = wave; h < NH; h += 8) {
    float* row = sc + h * SEQ;
    float m = -1e30f;
    for (int s = lane; s < SEQ; s += 32) m = fmaxf(m, row[s]);
#pragma unroll
    for (int off = 16; off > 0; off >>= 1) m = fmaxf(m, __shfl_xor(m, off, 32));
    float sum = 0.f;
    for (int s = lane; s < SEQ; s += 32) {
      const float e = __expf(row[s] - m);
      row[s] = e;                       // this wave exclusively owns this row
      sum += e;
    }
#pragma unroll
    for (int off = 16; off > 0; off >>= 1) sum += __shfl_xor(sum, off, 32);
    if (lane == 0) rl[h] = 1.f / sum;
  }
  __syncthreads();

  // normalize + head mix + write bf16
  bf16_t* __restrict__ dst =
      pmix + ((size_t)b * NH) * (size_t)T_LEN * SEQ + (size_t)t * SEQ;
  for (int s = threadIdx.x; s < SEQ; s += 256) {
    float e[NH];
#pragma unroll
    for (int h = 0; h < NH; ++h) e[h] = sc[h * SEQ + s] * rl[h];
#pragma unroll
    for (int g = 0; g < NH; ++g) {
      float a = 0.f;
#pragma unroll
      for (int h = 0; h < NH; ++h) a += Wm[h * NH + g] * e[h];
      dst[(size_t)g * T_LEN * SEQ + s] = (bf16_t)a;
    }
  }
}

// ======================================================================
// Kernel 4: ctx[b,g,t,d] = sum_s pmix[b,g,t,s] * vT[b,g,d,s]  (NT, bf16)
// A rows feed two waves (N=64 is two 32-wide tile columns): register-tiled 32x32.
// ======================================================================
__global__ __launch_bounds__(256) void ctx_kernel(
    const bf16_t* __restrict__ pmix, const bf16_t* __restrict__ vt,
    bf16_t* __restrict__ ctxm) {
  const int lane = threadIdx.x & 31;
  const int wave = threadIdx.x >> 5;
  const int gtile = blockIdx.x * 8 + wave;         // 32*64 = 2048 tiles
  const int bh   = gtile >> 6;
  const int tile = gtile & 63;
  const int tm = tile >> 1;
  const int tn = tile & 1;
  const int r16  = lane & 15;
  const int koff = (lane & 16) ? 8 : 0;

  const bf16_t* __restrict__ A = pmix + (size_t)bh * T_LEN * SEQ;
  const bf16_t* __restrict__ B = vt + (size_t)bh * HD * T_LEN;
  const bf16_t* __restrict__ pa0 = A + (size_t)(tm * 32 + r16) * SEQ + koff;
  const bf16_t* __restrict__ pa1 = pa0 + (size_t)16 * SEQ;
  const bf16_t* __restrict__ pb0 = B + (size_t)(tn * 32 + r16) * T_LEN + koff;
  const bf16_t* __restrict__ pb1 = pb0 + (size_t)16 * T_LEN;

  v8f acc00 = {}, acc01 = {}, acc10 = {}, acc11 = {};
#pragma unroll 2
  for (int kb = 0; kb < SEQ; kb += 32) {
    __builtin_prefetch(pa0 + kb + 128, 0, 3);
    __builtin_prefetch(pb0 + kb + 128, 0, 3);
    v16bf a0 = frag_bf16(pa0 + kb);
    v16bf a1 = frag_bf16(pa1 + kb);
    v16bf b0 = frag_bf16(pb0 + kb);
    v16bf b1 = frag_bf16(pb1 + kb);
    acc00 = wmma_bf16(a0, b0, acc00);
    acc01 = wmma_bf16(a0, b1, acc01);
    acc10 = wmma_bf16(a1, b0, acc10);
    acc11 = wmma_bf16(a1, b1, acc11);
  }

  const int bb = bh >> 4;
  const int g  = bh & 15;
  v8f accs[2][2] = {{acc00, acc01}, {acc10, acc11}};
#pragma unroll
  for (int ti = 0; ti < 2; ++ti) {
#pragma unroll
    for (int tj = 0; tj < 2; ++tj) {
      const int d = tn * 32 + tj * 16 + r16;
      const int mbase = tm * 32 + ti * 16 + ((lane >> 4) << 3);
#pragma unroll
      for (int i = 0; i < 8; ++i) {
        const int t = mbase + i;
        ctxm[((size_t)t * BSZ + bb) * EMB + g * HD + d] =
            (bf16_t)accs[ti][tj][i];
      }
    }
  }
}

// ======================================================================
// Kernel 5: out = ctxm @ wo^T + bo  (NT, bf16, LDS-tiled like proj)
// ======================================================================
__global__ __launch_bounds__(256) void out_kernel(
    const bf16_t* __restrict__ ctxm, const bf16_t* __restrict__ wo,
    const float* __restrict__ bo, float* __restrict__ out) {
  __shared__ __align__(16) bf16_t Asm[2][BM * LDSTRIDE];
  __shared__ __align__(16) bf16_t Bsm[2][BN * LDSTRIDE];

  const int tid  = threadIdx.x;
  const int lane = tid & 31;
  const int wave = tid >> 5;
  const int wm = wave >> 1;
  const int wn = wave & 1;
  const int bm = blockIdx.x >> 3;
  const int bn = blockIdx.x & 7;
  const int r16  = lane & 15;
  const int koff = (lane & 16) ? 8 : 0;

  const bf16_t* __restrict__ gA = ctxm + (size_t)(bm * BM) * EMB;
  const bf16_t* __restrict__ gB = wo + (size_t)(bn * BN) * EMB;

  v8f acc[2][4] = {};
  stage_tile_async(gA, EMB, 0, Asm[0], tid);
  stage_tile_async(gB, EMB, 0, Bsm[0], tid);

  const int NK = EMB / BK;
  for (int it = 0; it < NK; ++it) {
    const int cur = it & 1;
    if (it + 1 < NK) {
      stage_tile_async(gA, EMB, (it + 1) * BK, Asm[cur ^ 1], tid);
      stage_tile_async(gB, EMB, (it + 1) * BK, Bsm[cur ^ 1], tid);
      asm volatile("s_wait_asynccnt 0x4" ::: "memory");
    } else {
      asm volatile("s_wait_asynccnt 0x0" ::: "memory");
    }
    __syncthreads();

    const bf16_t* Ab = Asm[cur] + (wm * 32) * LDSTRIDE;
    const bf16_t* Bb = Bsm[cur] + (wn * 64) * LDSTRIDE;
    v16bf a0 = frag_bf16(Ab + (size_t)r16 * LDSTRIDE + koff);
    v16bf a1 = frag_bf16(Ab + (size_t)(16 + r16) * LDSTRIDE + koff);
    v16bf b0 = frag_bf16(Bb + (size_t)r16 * LDSTRIDE + koff);
    v16bf b1 = frag_bf16(Bb + (size_t)(16 + r16) * LDSTRIDE + koff);
    v16bf b2 = frag_bf16(Bb + (size_t)(32 + r16) * LDSTRIDE + koff);
    v16bf b3 = frag_bf16(Bb + (size_t)(48 + r16) * LDSTRIDE + koff);
    acc[0][0] = wmma_bf16(a0, b0, acc[0][0]);
    acc[0][1] = wmma_bf16(a0, b1, acc[0][1]);
    acc[0][2] = wmma_bf16(a0, b2, acc[0][2]);
    acc[0][3] = wmma_bf16(a0, b3, acc[0][3]);
    acc[1][0] = wmma_bf16(a1, b0, acc[1][0]);
    acc[1][1] = wmma_bf16(a1, b1, acc[1][1]);
    acc[1][2] = wmma_bf16(a1, b2, acc[1][2]);
    acc[1][3] = wmma_bf16(a1, b3, acc[1][3]);
    __syncthreads();
  }

#pragma unroll
  for (int ti = 0; ti < 2; ++ti) {
#pragma unroll
    for (int tj = 0; tj < 4; ++tj) {
      const int n = bn * BN + wn * 64 + tj * 16 + r16;
      const float bv = bo[n];
      const int mbase = bm * BM + wm * 32 + ti * 16 + ((lane >> 4) << 3);
#pragma unroll
      for (int i = 0; i < 8; ++i) {
        const int m = mbase + i;
        out[(size_t)m * EMB + n] = acc[ti][tj][i] + bv;
      }
    }
  }
}

// ======================================================================
// launch
// ======================================================================
extern "C" void kernel_launch(void* const* d_in, const int* in_sizes, int n_in,
                              void* d_out, int out_size, void* d_ws, size_t ws_size,
                              hipStream_t stream) {
  const float* query = (const float*)d_in[0];
  const float* wq    = (const float*)d_in[1];
  const float* bq    = (const float*)d_in[2];
  const float* wk    = (const float*)d_in[3];
  const float* bk    = (const float*)d_in[4];
  const float* wv    = (const float*)d_in[5];
  const float* bv    = (const float*)d_in[6];
  const float* wo    = (const float*)d_in[7];
  const float* bo    = (const float*)d_in[8];
  const float* mix_w = (const float*)d_in[9];
  float* out = (float*)d_out;

  // workspace carve-up (~220 MB; essentially L2-resident on MI455X)
  char* ws = (char*)d_ws;
  bf16_t* qb     = (bf16_t*)(ws);                        //  4 MB [MROWS,EMB]
  bf16_t* wqb    = (bf16_t*)(ws + (4ull   << 20));       //  2 MB
  bf16_t* wkb    = (bf16_t*)(ws + (6ull   << 20));       //  2 MB
  bf16_t* wvb    = (bf16_t*)(ws + (8ull   << 20));       //  2 MB
  bf16_t* wob    = (bf16_t*)(ws + (10ull  << 20));       //  2 MB
  bf16_t* qh     = (bf16_t*)(ws + (12ull  << 20));       //  4 MB [B,H,T,D]
  bf16_t* kh     = (bf16_t*)(ws + (16ull  << 20));       //  4 MB [B,H,T,D]
  bf16_t* vt     = (bf16_t*)(ws + (20ull  << 20));       //  4 MB [B,H,D,T]
  bf16_t* ctxm   = (bf16_t*)(ws + (24ull  << 20));       //  4 MB [T,B,E]
  float*  scores = (float*) (ws + (28ull  << 20));       // 128 MB [B,H,T,S]
  bf16_t* pmix   = (bf16_t*)(ws + (156ull << 20));       //  64 MB [B,H,T,S]

  const dim3 blk(256);
  // one-shot bf16 conversion of activations and weights
  conv_kernel<<<1024, blk, 0, stream>>>(query, qb);
  conv_kernel<<<512,  blk, 0, stream>>>(wq, wqb);
  conv_kernel<<<512,  blk, 0, stream>>>(wk, wkb);
  conv_kernel<<<512,  blk, 0, stream>>>(wv, wvb);
  conv_kernel<<<512,  blk, 0, stream>>>(wo, wob);
  // Q/K/V projections (LDS-tiled; Q pre-scaled; V stored transposed)
  proj_kernel<<<128, blk, 0, stream>>>(qb, wqb, bq, qh, QSCALE, 0);
  proj_kernel<<<128, blk, 0, stream>>>(qb, wkb, bk, kh, 1.0f, 0);
  proj_kernel<<<128, blk, 0, stream>>>(qb, wvb, bv, vt, 1.0f, 1);
  // scores
  scores_kernel<<<4096, blk, 0, stream>>>(qh, kh, scores);
  // softmax + head mix (dynamic LDS: 16 rows f32 + mix matrix + stats)
  const size_t smem = (size_t)(NH * SEQ + NH * NH + NH) * sizeof(float);
  softmax_mix_kernel<<<T_LEN * BSZ, blk, smem, stream>>>(scores, mix_w, pmix);
  // probs @ V, merge heads
  ctx_kernel<<<256, blk, 0, stream>>>(pmix, vt, ctxm);
  // output projection (LDS-tiled)
  out_kernel<<<128, blk, 0, stream>>>(ctxm, wob, bo, out);
}
